// GNNSegmentClassifier_28630251995176
// MI455X (gfx1250) — matrix operand order repaired and checked by reading, not compiled, not run on hardware
//
#include <hip/hip_runtime.h>
#include <hip/hip_bf16.h>

typedef _Float16 v8h  __attribute__((ext_vector_type(8)));
typedef _Float16 v16h __attribute__((ext_vector_type(16)));
typedef float    v8f  __attribute__((ext_vector_type(8)));

#define WPB 8      // waves per block
#define TPB 256

static __device__ __forceinline__ v8f wmma16(v16h a, v16h b, v8f c) {
    // D = A(16x32 f16) x B(32x16 f16) + C(16x16 f32)
    return __builtin_amdgcn_wmma_f32_16x16x32_f16(
        /*neg_a=*/false, a, /*neg_b=*/false, b,
        /*c_mod=*/(short)0, c, /*reuse_a=*/false, /*reuse_b=*/false);
}

// Branch-free activations: single v_exp_f32 + v_rcp_f32, no EXEC divergence.
static __device__ __forceinline__ float tanh_fast(float x) {
    // tanh(x) = 1 - 2/(1 + exp(2x));  2*log2(e) = 2.8853900817779268
    const float e = __builtin_amdgcn_exp2f(x * 2.8853900817779268f);
    return 1.f - 2.f * __builtin_amdgcn_rcpf(1.f + e);
}
static __device__ __forceinline__ float sigmoid_fast(float x) {
    // 1/(1 + exp(-x));  log2(e) = 1.4426950408889634
    const float e = __builtin_amdgcn_exp2f(-x * 1.4426950408889634f);
    return __builtin_amdgcn_rcpf(1.f + e);
}

// B-fragment for We1 (rows 0..69 of [70,32]) with A-column mapping:
// A col k<35 -> We1 row k (bo part); 40<=k<75 -> We1 row k-5 (bi part); else 0.
static __device__ __forceinline__ v16h bfrag_we1(const float* W, int s, int t, int lane) {
    const int col   = t * 16 + (lane & 15);
    const int kbase = s * 32 + ((lane & 16) ? 8 : 0);
    v16h f;
#pragma unroll
    for (int j = 0; j < 16; ++j) {
        const int k = kbase + (j < 8 ? j : j + 8);
        float v = 0.f;
        if (k < 35)                 v = W[k * 32 + col];
        else if (k >= 40 && k < 75) v = W[(k - 5) * 32 + col];
        f[j] = (_Float16)v;
    }
    return f;
}

// B-fragment for a dense [K,32] row-major weight, zero-padded past K.
static __device__ __forceinline__ v16h bfrag_dense(const float* W, int K, int s, int t, int lane) {
    const int col   = t * 16 + (lane & 15);
    const int kbase = s * 32 + ((lane & 16) ? 8 : 0);
    v16h f;
#pragma unroll
    for (int j = 0; j < 16; ++j) {
        const int k = kbase + (j < 8 ? j : j + 8);
        f[j] = (_Float16)((k < K) ? W[k * 32 + col] : 0.f);
    }
    return f;
}

// A-fragment (16x32 f16) from a row-major LDS tile with row stride `stride` halfs.
// ISA layout: lane<16 row=lane holds K ks+0..7 / ks+16..23; lane>=16 row=lane-16 holds +8.
static __device__ __forceinline__ v16h afrag(const _Float16* tile, int stride, int slot, int side, int s) {
    const v8h* p = (const v8h*)(tile + slot * stride + s * 32 + side * 8);
    const v8h lo = p[0];
    const v8h hi = p[2];
    v16h a;
#pragma unroll
    for (int j = 0; j < 8; ++j) { a[j] = lo[j]; a[8 + j] = hi[j]; }
    return a;
}

// ---------------- input network: H = [tanh(X@W_in+b_in) | X | 0pad] as f16[N,40]
__global__ void input_kernel(const float* __restrict__ X,
                             const float* __restrict__ W_in, const float* __restrict__ b_in,
                             _Float16* __restrict__ Ha, _Float16* __restrict__ Hb, int N) {
    const int n = blockIdx.x * blockDim.x + threadIdx.x;
    if (n >= N) return;
    const float x0 = X[n * 3 + 0], x1 = X[n * 3 + 1], x2 = X[n * 3 + 2];
    _Float16* ra = Ha + (size_t)n * 40;
    _Float16* rb = Hb + (size_t)n * 40;
#pragma unroll
    for (int c = 0; c < 32; ++c) {
        const float v = tanh_fast(x0 * W_in[c] + x1 * W_in[32 + c] + x2 * W_in[64 + c] + b_in[c]);
        ra[c] = (_Float16)v;
    }
    ra[32] = (_Float16)x0; ra[33] = (_Float16)x1; ra[34] = (_Float16)x2;
    rb[32] = ra[32];       rb[33] = ra[33];       rb[34] = ra[34];
#pragma unroll
    for (int c = 35; c < 40; ++c) { ra[c] = (_Float16)0.f; rb[c] = (_Float16)0.f; }
}

// ---------------- edge network (+ optional message scatter)
__global__ __launch_bounds__(TPB) void edge_kernel(
    const _Float16* __restrict__ H, const int* __restrict__ send, const int* __restrict__ recv,
    const float* __restrict__ We1, const float* __restrict__ be1,
    const float* __restrict__ We2, const float* __restrict__ be2,
    float* __restrict__ mi, float* __restrict__ mo, float* __restrict__ out,
    int E, int writeOut) {
    __shared__ __align__(16) _Float16 Asmem[WPB][16 * 96];  // 16 edges x K(=96 padded)
    __shared__ float Hsmem[WPB][16 * 33];                   // 16 edges x 32 hidden
    const int wave = threadIdx.x >> 5;
    const int lane = threadIdx.x & 31;
    _Float16* At = Asmem[wave];
    float*    Ht = Hsmem[wave];

    // zero K-pad region (cols 80..95) once; never overwritten in the loop
    for (int i = lane; i < 16 * 16; i += 32) At[(i >> 4) * 96 + 80 + (i & 15)] = (_Float16)0.f;

    v16h B[3][2];
#pragma unroll
    for (int s = 0; s < 3; ++s) {
        B[s][0] = bfrag_we1(We1, s, 0, lane);
        B[s][1] = bfrag_we1(We1, s, 1, lane);
    }
    const float b1a = be1[lane & 15];
    const float b1b = be1[16 + (lane & 15)];
    float w2[32];
#pragma unroll
    for (int j = 0; j < 32; ++j) w2[j] = We2[j];
    const float b2 = be2[0];

    const int slot = lane & 15;   // edge slot within tile
    const int side = lane >> 4;   // 0 = send row (bo), 1 = recv row (bi)
    const int nTiles = E >> 4;
    const int waveId = blockIdx.x * WPB + wave;
    const int nWaves = gridDim.x * WPB;

    for (int t = waveId; t < nTiles; t += nWaves) {
        const int e  = (t << 4) + slot;
        const int si = send[e];
        const int ri = recv[e];
        const int gnode = side ? ri : si;
        // gather one 40-half H row (80B, 16B aligned) into the A tile
        const uint4* src = (const uint4*)(H + (size_t)gnode * 40);
        uint4* dst = (uint4*)(At + slot * 96 + side * 40);
#pragma unroll
        for (int j = 0; j < 5; ++j) dst[j] = src[j];
        asm volatile("s_wait_dscnt 0" ::: "memory");

        v8f a0 = {}, a1 = {};
#pragma unroll
        for (int s = 0; s < 3; ++s) {
            const v16h a = afrag(At, 96, slot, side, s);
            a0 = wmma16(a, B[s][0], a0);
            a1 = wmma16(a, B[s][1], a1);
        }
        // h = tanh(pre + be1) -> LDS (C layout: lane<16 rows 0..7, lane>=16 rows 8..15)
        const int col = lane & 15, rb = side * 8;
#pragma unroll
        for (int r = 0; r < 8; ++r) {
            Ht[(rb + r) * 33 + col]      = tanh_fast(a0[r] + b1a);
            Ht[(rb + r) * 33 + col + 16] = tanh_fast(a1[r] + b1b);
        }
        asm volatile("s_wait_dscnt 0" ::: "memory");
        // e = sigmoid(h . We2 + be2); both half-waves compute their slot's edge
        float d = b2;
#pragma unroll
        for (int j = 0; j < 32; ++j) d += Ht[slot * 33 + j] * w2[j];
        const float ev = sigmoid_fast(d);

        if (writeOut) {
            if (side == 0) out[e] = ev;
        } else {
            // side0 holds bo=H[send] -> mi[recv]; side1 holds bi=H[recv] -> mo[send]
            float* dstp = side ? (mo + (size_t)si * 35) : (mi + (size_t)ri * 35);
            const _Float16* feat = At + slot * 96 + side * 40;
#pragma unroll
            for (int k = 0; k < 35; ++k) atomicAdd(dstp + k, ev * (float)feat[k]);
        }
    }
}

// ---------------- node network: H' = [tanh(tanh([mi|mo|H]@Wn1+bn1)@Wn2+bn2) | X]
__global__ __launch_bounds__(TPB) void node_kernel(
    const float* __restrict__ mi, const float* __restrict__ mo,
    const _Float16* __restrict__ Hin, _Float16* __restrict__ Hout,
    const float* __restrict__ Wn1, const float* __restrict__ bn1,
    const float* __restrict__ Wn2, const float* __restrict__ bn2, int N) {
    __shared__ __align__(16) _Float16 Asmem[WPB][16 * 128];  // 16 nodes x K(=128 padded)
    __shared__ __align__(16) _Float16 A2smem[WPB][16 * 32];  // hidden tile for 2nd GEMM
    const int wave = threadIdx.x >> 5;
    const int lane = threadIdx.x & 31;
    _Float16* At = Asmem[wave];
    _Float16* A2 = A2smem[wave];

    for (int i = lane; i < 16 * 23; i += 32) At[(i / 23) * 128 + 105 + (i % 23)] = (_Float16)0.f;

    v16h B1[4][2];
#pragma unroll
    for (int s = 0; s < 4; ++s) {
        B1[s][0] = bfrag_dense(Wn1, 105, s, 0, lane);
        B1[s][1] = bfrag_dense(Wn1, 105, s, 1, lane);
    }
    v16h B2[2];
    B2[0] = bfrag_dense(Wn2, 32, 0, 0, lane);
    B2[1] = bfrag_dense(Wn2, 32, 0, 1, lane);
    const float b1a = bn1[lane & 15], b1b = bn1[16 + (lane & 15)];
    const float b2a = bn2[lane & 15], b2b = bn2[16 + (lane & 15)];

    const int slot = lane & 15, part = lane >> 4;
    const int nTiles = N >> 4;  // N divisible by 16
    const int waveId = blockIdx.x * WPB + wave;
    const int nWaves = gridDim.x * WPB;

    for (int t = waveId; t < nTiles; t += nWaves) {
        const int node = (t << 4) + slot;
        _Float16* arow = At + slot * 128;
        const float* msrc = part ? (mo + (size_t)node * 35) : (mi + (size_t)node * 35);
        const int cb = part ? 35 : 0;
#pragma unroll
        for (int k = 0; k < 35; ++k) arow[cb + k] = (_Float16)msrc[k];
        // interleaved copy of H (35 halfs) split across the two half-waves
        const _Float16* hs = Hin + (size_t)node * 40;
#pragma unroll
        for (int k = 0; k < 18; ++k) {
            const int c = 2 * k + part;
            if (c < 35) arow[70 + c] = hs[c];
        }
        asm volatile("s_wait_dscnt 0" ::: "memory");

        v8f a0 = {}, a1 = {};
#pragma unroll
        for (int s = 0; s < 4; ++s) {
            const v16h a = afrag(At, 128, slot, part, s);
            a0 = wmma16(a, B1[s][0], a0);
            a1 = wmma16(a, B1[s][1], a1);
        }
        const int col = lane & 15, rb = part * 8;
#pragma unroll
        for (int r = 0; r < 8; ++r) {
            A2[(rb + r) * 32 + col]      = (_Float16)tanh_fast(a0[r] + b1a);
            A2[(rb + r) * 32 + col + 16] = (_Float16)tanh_fast(a1[r] + b1b);
        }
        asm volatile("s_wait_dscnt 0" ::: "memory");

        const v16h a2 = afrag(A2, 32, slot, part, 0);
        v8f c0 = {}, c1 = {};
        c0 = wmma16(a2, B2[0], c0);
        c1 = wmma16(a2, B2[1], c1);
        // write new hidden (cols 0..31); X/pad cols persist from input_kernel
#pragma unroll
        for (int r = 0; r < 8; ++r) {
            const size_t nr = (size_t)((t << 4) + rb + r) * 40;
            Hout[nr + col]      = (_Float16)tanh_fast(c0[r] + b2a);
            Hout[nr + col + 16] = (_Float16)tanh_fast(c1[r] + b2b);
        }
    }
}

extern "C" void kernel_launch(void* const* d_in, const int* in_sizes, int n_in,
                              void* d_out, int out_size, void* d_ws, size_t ws_size,
                              hipStream_t stream) {
    const float* X    = (const float*)d_in[0];
    const int*   send = (const int*)d_in[1];
    const int*   recv = (const int*)d_in[2];
    const float* W_in = (const float*)d_in[3];
    const float* b_in = (const float*)d_in[4];
    const float* We1  = (const float*)d_in[5];
    const float* be1  = (const float*)d_in[6];
    const float* We2  = (const float*)d_in[7];
    const float* be2  = (const float*)d_in[8];
    const float* Wn1  = (const float*)d_in[9];
    const float* bn1  = (const float*)d_in[10];
    const float* Wn2  = (const float*)d_in[11];
    const float* bn2  = (const float*)d_in[12];

    const int N = in_sizes[0] / 3;
    const int E = in_sizes[1];

    char* ws = (char*)d_ws;
    const size_t szH = (((size_t)N * 40 * sizeof(_Float16)) + 255) & ~(size_t)255;
    const size_t szM = (((size_t)N * 35 * sizeof(float)) + 255) & ~(size_t)255;
    _Float16* Ha = (_Float16*)ws;
    _Float16* Hb = (_Float16*)(ws + szH);
    float*    mi = (float*)(ws + 2 * szH);
    float*    mo = (float*)(ws + 2 * szH + szM);
    float*    out = (float*)d_out;
    if (ws_size < 2 * szH + 2 * szM) return;

    input_kernel<<<(N + TPB - 1) / TPB, TPB, 0, stream>>>(X, W_in, b_in, Ha, Hb, N);

    _Float16* Hcur = Ha;
    _Float16* Hnxt = Hb;
    for (int it = 0; it < 4; ++it) {
        hipMemsetAsync(mi, 0, 2 * szM, stream);  // mi and mo are contiguous
        edge_kernel<<<1024, TPB, 0, stream>>>(Hcur, send, recv, We1, be1, We2, be2,
                                              mi, mo, out, E, 0);
        node_kernel<<<256, TPB, 0, stream>>>(mi, mo, Hcur, Hnxt, Wn1, bn1, Wn2, bn2, N);
        _Float16* tmp = Hcur; Hcur = Hnxt; Hnxt = tmp;
    }
    edge_kernel<<<1024, TPB, 0, stream>>>(Hcur, send, recv, We1, be1, We2, be2,
                                          mi, mo, out, E, 1);
}
